// PerformerAttention_14061722927424
// MI455X (gfx1250) — matrix operand order repaired and checked by reading, not compiled, not run on hardware
//
#include <hip/hip_runtime.h>
#include <hip/hip_bf16.h>
#include <math.h>

// ---------------------------------------------------------------------------
// Performer (FAVOR+) attention for MI455X / gfx1250, bf16 WMMA compute.
//   B=4, L=4096, D=1024, H=16, DH=64, M=266 random features (padded to 272).
// GEMM stages: v_wmma_f32_16x16x32_bf16, 128x64x32 tiles, 8 wave32s/WG,
// register-double-buffered global->LDS staging with b128 loads.
// ---------------------------------------------------------------------------

#define BATCH   4
#define SEQ     4096
#define DIM     1024
#define HEADS   16
#define DHEAD   64
#define MFEAT   266
#define MPAD    272          // 266 rounded up to a multiple of 16
#define BL      (BATCH * SEQ)        // 16384 token rows
#define BH      (BATCH * HEADS)     // 64 (b,h) batches

typedef __attribute__((ext_vector_type(16))) __bf16 v16bf;
typedef __attribute__((ext_vector_type(4)))  __bf16 v4bf;
typedef __attribute__((ext_vector_type(8)))  float  v8f;

// ---------------------------------------------------------------------------
// Generic batched WMMA GEMM:  C = alpha * (op(A) * op(B) + bias)
//   TA=false: A is M x K row-major                     A[m*lda + k]
//   TA=true : A stored K x M row-major (A^T logically) A[k*lda + m]
//   TB=false: B stored N x K row-major (weight layout) B[n*ldb + k]
//   TB=true : B stored K x N row-major                 B[k*ldb + n]
// Batch z offset per operand: off(z) = (z/inner)*hi + (z%inner)*lo
// Tile: 128x64x32; 256 threads = 8 wave32s; each wave: 16 rows x 64 cols
// (4 accumulators reusing one A fragment).
// ---------------------------------------------------------------------------
template <bool TA, bool TB>
__global__ __launch_bounds__(256) void wmma_gemm(
    const float* __restrict__ A, long lda, long aHi, long aLo, int aInner,
    const float* __restrict__ Bm, long ldb, long bHi, long bLo, int bInner,
    float*       __restrict__ C, long ldc, long cHi, long cLo, int cInner,
    const float* __restrict__ bias, float alpha,
    int Mdim, int Ndim, int Kdim)
{
    const int z = blockIdx.z;
    A  += (long)(z / aInner) * aHi + (long)(z % aInner) * aLo;
    Bm += (long)(z / bInner) * bHi + (long)(z % bInner) * bLo;
    C  += (long)(z / cInner) * cHi + (long)(z % cInner) * cLo;

    const int m0 = blockIdx.y * 128;
    const int n0 = blockIdx.x * 64;

    __shared__ alignas(16) __bf16 As[128][40];  // [m][k], 80B row stride
    __shared__ alignas(16) __bf16 Bs[64][40];   // [n][k]

    const int tid  = threadIdx.x;
    const int wave = tid >> 5;       // 0..7 -> 16-row M sub-tile
    const int lane = tid & 31;
    const int r    = lane & 15;
    const int g    = lane >> 4;

    // staging registers: A tile 128x32 -> 16 floats/thread, B 64x32 -> 8
    float4 ra[4];
    float4 rb[2];
    float* rap = (float*)ra;
    float* rbp = (float*)rb;

    auto loadTiles = [&](int kb) {
        const bool aFull = (m0 + 128 <= Mdim) && (kb + 32 <= Kdim);
        const bool bFull = (n0 + 64  <= Ndim) && (kb + 32 <= Kdim);
        if (!TA) {
            if (aFull) {
                #pragma unroll
                for (int c = 0; c < 4; ++c) {
                    const int idx = tid + c * 256;
                    const int row = idx >> 3, kq = (idx & 7) * 4;
                    ra[c] = *(const float4*)&A[(long)(m0 + row) * lda + kb + kq];
                }
            } else {
                #pragma unroll
                for (int c = 0; c < 4; ++c) {
                    const int idx = tid + c * 256;
                    const int row = idx >> 3, kq = (idx & 7) * 4;
                    #pragma unroll
                    for (int i = 0; i < 4; ++i)
                        rap[c * 4 + i] = (m0 + row < Mdim && kb + kq + i < Kdim)
                            ? A[(long)(m0 + row) * lda + kb + kq + i] : 0.f;
                }
            }
        } else {
            if (aFull) {
                #pragma unroll
                for (int c = 0; c < 4; ++c) {
                    const int idx = tid + c * 256;
                    const int kk = idx >> 5, mq = (idx & 31) * 4;
                    ra[c] = *(const float4*)&A[(long)(kb + kk) * lda + m0 + mq];
                }
            } else {
                #pragma unroll
                for (int c = 0; c < 4; ++c) {
                    const int idx = tid + c * 256;
                    const int kk = idx >> 5, mq = (idx & 31) * 4;
                    #pragma unroll
                    for (int i = 0; i < 4; ++i)
                        rap[c * 4 + i] = (m0 + mq + i < Mdim && kb + kk < Kdim)
                            ? A[(long)(kb + kk) * lda + m0 + mq + i] : 0.f;
                }
            }
        }
        if (!TB) {
            if (bFull) {
                #pragma unroll
                for (int c = 0; c < 2; ++c) {
                    const int idx = tid + c * 256;
                    const int row = idx >> 3, kq = (idx & 7) * 4;
                    rb[c] = *(const float4*)&Bm[(long)(n0 + row) * ldb + kb + kq];
                }
            } else {
                #pragma unroll
                for (int c = 0; c < 2; ++c) {
                    const int idx = tid + c * 256;
                    const int row = idx >> 3, kq = (idx & 7) * 4;
                    #pragma unroll
                    for (int i = 0; i < 4; ++i)
                        rbp[c * 4 + i] = (n0 + row < Ndim && kb + kq + i < Kdim)
                            ? Bm[(long)(n0 + row) * ldb + kb + kq + i] : 0.f;
                }
            }
        } else {
            if (bFull) {
                #pragma unroll
                for (int c = 0; c < 2; ++c) {
                    const int idx = tid + c * 256;
                    const int kk = idx >> 4, nq = (idx & 15) * 4;
                    rb[c] = *(const float4*)&Bm[(long)(kb + kk) * ldb + n0 + nq];
                }
            } else {
                #pragma unroll
                for (int c = 0; c < 2; ++c) {
                    const int idx = tid + c * 256;
                    const int kk = idx >> 4, nq = (idx & 15) * 4;
                    #pragma unroll
                    for (int i = 0; i < 4; ++i)
                        rbp[c * 4 + i] = (n0 + nq + i < Ndim && kb + kk < Kdim)
                            ? Bm[(long)(kb + kk) * ldb + n0 + nq + i] : 0.f;
                }
            }
        }
    };

    auto storeTiles = [&]() {
        if (!TA) {
            #pragma unroll
            for (int c = 0; c < 4; ++c) {
                const int idx = tid + c * 256;
                const int row = idx >> 3, kq = (idx & 7) * 4;
                v4bf p;
                #pragma unroll
                for (int i = 0; i < 4; ++i) p[i] = (__bf16)rap[c * 4 + i];
                *(v4bf*)&As[row][kq] = p;      // 8B-aligned (row stride 80B)
            }
        } else {
            #pragma unroll
            for (int c = 0; c < 4; ++c) {
                const int idx = tid + c * 256;
                const int kk = idx >> 5, mq = (idx & 31) * 4;
                #pragma unroll
                for (int i = 0; i < 4; ++i)
                    As[mq + i][kk] = (__bf16)rap[c * 4 + i];
            }
        }
        if (!TB) {
            #pragma unroll
            for (int c = 0; c < 2; ++c) {
                const int idx = tid + c * 256;
                const int row = idx >> 3, kq = (idx & 7) * 4;
                v4bf p;
                #pragma unroll
                for (int i = 0; i < 4; ++i) p[i] = (__bf16)rbp[c * 4 + i];
                *(v4bf*)&Bs[row][kq] = p;
            }
        } else {
            #pragma unroll
            for (int c = 0; c < 2; ++c) {
                const int idx = tid + c * 256;
                const int kk = idx >> 4, nq = (idx & 15) * 4;
                #pragma unroll
                for (int i = 0; i < 4; ++i)
                    Bs[nq + i][kk] = (__bf16)rbp[c * 4 + i];
            }
        }
    };

    v8f acc[4] = {};

    // prologue: stage first tile
    loadTiles(0);
    storeTiles();
    __syncthreads();

    for (int k0 = 0; k0 < Kdim; k0 += 32) {
        const bool more = (k0 + 32) < Kdim;
        if (more) loadTiles(k0 + 32);      // overlap with WMMA below

        // ---- A fragment (ISA 16-bit A 16x32 layout) ----
        v16bf afrag;
        #pragma unroll
        for (int t = 0; t < 16; ++t) {
            const int j = t >> 1, half = t & 1;
            const int kk = ((j >= 4) ? 16 : 0) + g * 8 + (j & 3) * 2 + half;
            afrag[t] = As[wave * 16 + r][kk];
        }
        // ---- 4 B fragments / 4 WMMAs reusing afrag ----
        #pragma unroll
        for (int nt = 0; nt < 4; ++nt) {
            v16bf bfrag;
            #pragma unroll
            for (int t = 0; t < 16; ++t)
                bfrag[t] = Bs[nt * 16 + r][((t >= 8) ? 16 : 0) + g * 8 + (t & 7)];
            acc[nt] = __builtin_amdgcn_wmma_f32_16x16x32_bf16(
                false, afrag, false, bfrag, (short)0, acc[nt], false, false);
        }

        __syncthreads();
        if (more) { storeTiles(); }
        __syncthreads();
    }

    // ---- epilogue: C/D layout -> VGPR i: M = i + 8*g, N = lane&15 ----
    #pragma unroll
    for (int nt = 0; nt < 4; ++nt) {
        const int nn = n0 + nt * 16 + r;
        if (nn < Ndim) {
            const float bv = bias ? bias[nn] : 0.f;
            #pragma unroll
            for (int i = 0; i < 8; ++i) {
                const int mm = m0 + wave * 16 + i + g * 8;
                if (mm < Mdim)
                    C[(long)mm * ldc + nn] = alpha * (acc[nt][i] + bv);
            }
        }
    }
}

// ---------------------------------------------------------------------------
// phi: FAVOR+ exp feature map, in-place on the raw projection rows.
//   row <- exp(-0.5*||x||^2 + proj - max_m(proj)) / sqrt(M); pad cols -> 0.
// One 64-thread block per (b,h,l) row.
// ---------------------------------------------------------------------------
__global__ __launch_bounds__(64) void phi_kernel(
    const float* __restrict__ xbase,   // (B*L, D) scaled head data
    float*       __restrict__ proj,    // (B*H, L, MPAD) raw projections
    float rsqrtM)
{
    const int rid = blockIdx.x;                 // 0 .. B*H*L-1
    const int l   = rid % SEQ;
    const int h   = (rid / SEQ) % HEADS;
    const int b   = rid / (SEQ * HEADS);
    const float* x   = xbase + ((long)(b * SEQ + l)) * DIM + h * DHEAD;
    float*       row = proj  + ((long)((b * HEADS + h) * SEQ + l)) * MPAD;

    __shared__ float red[64];
    const int t = threadIdx.x;

    // -0.5 * ||x||^2  (64 dims, one per thread)
    float s = x[t] * x[t];
    red[t] = s; __syncthreads();
    #pragma unroll
    for (int o = 32; o > 0; o >>= 1) { if (t < o) red[t] += red[t + o]; __syncthreads(); }
    const float hterm = -0.5f * red[0];
    __syncthreads();

    // row max over the 266 valid features (stabilizer)
    float mx = -3.0e38f;
    for (int c = t; c < MFEAT; c += 64) mx = fmaxf(mx, row[c]);
    red[t] = mx; __syncthreads();
    #pragma unroll
    for (int o = 32; o > 0; o >>= 1) { if (t < o) red[t] = fmaxf(red[t], red[t + o]); __syncthreads(); }
    const float stab = red[0];
    __syncthreads();

    for (int c = t; c < MPAD; c += 64) {
        float v = (c < MFEAT) ? expf(hterm + row[c] - stab) * rsqrtM : 0.f;
        row[c] = v;
    }
}

// ---------------------------------------------------------------------------
// ksum[z][m] = sum_l k'[z][l][m]    (one 128-thread block per (b,h))
// ---------------------------------------------------------------------------
__global__ __launch_bounds__(128) void ksum_kernel(
    const float* __restrict__ kp, float* __restrict__ ksum)
{
    const int z = blockIdx.x;
    const float* base = kp + (long)z * SEQ * MPAD;
    for (int c = threadIdx.x; c < MPAD; c += 128) {
        float s = 0.f;
        for (int l = 0; l < SEQ; ++l) s += base[(long)l * MPAD + c];
        ksum[(long)z * MPAD + c] = s;
    }
}

// ---------------------------------------------------------------------------
// denom + normalize + merge heads:
//   denom = q'[l] . ksum + 1e-6 ;  merged[b,l,h*64+d] = ctx_raw[z,l,d]/denom
// One wave32 per (z,l) row; 8 waves per block.
// ---------------------------------------------------------------------------
__global__ __launch_bounds__(256) void norm_merge_kernel(
    const float* __restrict__ qp,      // (BH, L, MPAD)
    const float* __restrict__ ksum,    // (BH, MPAD)
    const float* __restrict__ ctxraw,  // (BH, L, DHEAD)
    float*       __restrict__ merged)  // (B*L, D)
{
    const int lane = threadIdx.x & 31;
    const int rid  = blockIdx.x * 8 + (threadIdx.x >> 5);  // 0 .. BH*SEQ-1
    const int z = rid / SEQ;
    const int l = rid % SEQ;
    const int b = z / HEADS;
    const int h = z % HEADS;

    const float* qrow = qp + ((long)z * SEQ + l) * MPAD;
    const float* ks   = ksum + (long)z * MPAD;
    float s = 0.f;
    for (int c = lane; c < MPAD; c += 32) s += qrow[c] * ks[c];
    #pragma unroll
    for (int o = 16; o > 0; o >>= 1) s += __shfl_xor(s, o, 32);
    const float denom = s + 1e-6f;

    const float* cr = ctxraw + ((long)z * SEQ + l) * DHEAD;
    float* op = merged + ((long)(b * SEQ + l)) * DIM + h * DHEAD;
    for (int d = lane; d < DHEAD; d += 32) op[d] = cr[d] / denom;
}

// ---------------------------------------------------------------------------
// Host-side orchestration
// ---------------------------------------------------------------------------
extern "C" void kernel_launch(void* const* d_in, const int* in_sizes, int n_in,
                              void* d_out, int out_size, void* d_ws, size_t ws_size,
                              hipStream_t stream)
{
    const float* query = (const float*)d_in[0];
    const float* key   = (const float*)d_in[1];
    const float* value = (const float*)d_in[2];
    const float* q_w   = (const float*)d_in[3];
    const float* q_b   = (const float*)d_in[4];
    const float* k_w   = (const float*)d_in[5];
    const float* k_b   = (const float*)d_in[6];
    const float* v_w   = (const float*)d_in[7];
    const float* v_b   = (const float*)d_in[8];
    const float* out_w = (const float*)d_in[9];
    const float* out_b = (const float*)d_in[10];
    const float* rfeat = (const float*)d_in[11];   // (266, 64)
    float* out = (float*)d_out;

    // workspace layout (floats)
    float* qbuf  = (float*)d_ws;                       // (BL, D)   scaled q; later: merged ctx
    float* kbuf  = qbuf + (size_t)BL * DIM;            // (BL, D)   scaled k; later: ctx_raw
    float* vbuf  = kbuf + (size_t)BL * DIM;            // (BL, D)
    float* qp    = vbuf + (size_t)BL * DIM;            // (BH, L, MPAD) raw proj -> q'
    float* kp    = qp   + (size_t)BH * SEQ * MPAD;     // (BH, L, MPAD) raw proj -> k'
    float* kvb   = kp   + (size_t)BH * SEQ * MPAD;     // (BH, MPAD, DHEAD)
    float* ksumb = kvb  + (size_t)BH * MPAD * DHEAD;   // (BH, MPAD)

    const float scale  = 0.17677669529663687f;         // 1 / 1024^0.25
    const float rsqrtM = 0.06131393394849658f;         // 1 / sqrt(266)

    dim3 blk(256);
    #define GRID(Mv, Nv, Zv) dim3(((Nv) + 63) / 64, ((Mv) + 127) / 128, (Zv))

    // 1-3) input projections (alpha folds the 1/dim^0.25 scaling of q,k)
    wmma_gemm<false, false><<<GRID(BL, DIM, 1), blk, 0, stream>>>(
        query, DIM, 0, 0, 1,  q_w, DIM, 0, 0, 1,
        qbuf, DIM, 0, 0, 1,  q_b, scale, BL, DIM, DIM);
    wmma_gemm<false, false><<<GRID(BL, DIM, 1), blk, 0, stream>>>(
        key, DIM, 0, 0, 1,  k_w, DIM, 0, 0, 1,
        kbuf, DIM, 0, 0, 1,  k_b, scale, BL, DIM, DIM);
    wmma_gemm<false, false><<<GRID(BL, DIM, 1), blk, 0, stream>>>(
        value, DIM, 0, 0, 1,  v_w, DIM, 0, 0, 1,
        vbuf, DIM, 0, 0, 1,  v_b, 1.0f, BL, DIM, DIM);

    // 4-5) random-feature projections, per (b,h):  (L x 64) . (266 x 64)^T
    wmma_gemm<false, false><<<GRID(SEQ, MFEAT, BH), blk, 0, stream>>>(
        qbuf, DIM, (long)SEQ * DIM, DHEAD, HEADS,
        rfeat, DHEAD, 0, 0, 1,
        qp, MPAD, (long)SEQ * MPAD, 0, 1,
        nullptr, 1.0f, SEQ, MFEAT, DHEAD);
    wmma_gemm<false, false><<<GRID(SEQ, MFEAT, BH), blk, 0, stream>>>(
        kbuf, DIM, (long)SEQ * DIM, DHEAD, HEADS,
        rfeat, DHEAD, 0, 0, 1,
        kp, MPAD, (long)SEQ * MPAD, 0, 1,
        nullptr, 1.0f, SEQ, MFEAT, DHEAD);

    // 6-7) exp feature map (in place; writes exact zeros into pad columns)
    phi_kernel<<<BH * SEQ, 64, 0, stream>>>(qbuf, qp, rsqrtM);
    phi_kernel<<<BH * SEQ, 64, 0, stream>>>(kbuf, kp, rsqrtM);

    // 8) ksum[z][m] = sum_l k'
    ksum_kernel<<<BH, 128, 0, stream>>>(kp, ksumb);

    // 9) kv[z] = k'^T . v_head : (MPAD x L) . (L x DHEAD), K = 4096
    wmma_gemm<true, true><<<GRID(MPAD, DHEAD, BH), blk, 0, stream>>>(
        kp, MPAD, (long)SEQ * MPAD, 0, 1,
        vbuf, DIM, (long)SEQ * DIM, DHEAD, HEADS,
        kvb, DHEAD, (long)MPAD * DHEAD, 0, 1,
        nullptr, 1.0f, MPAD, DHEAD, SEQ);

    // 10) ctx_raw[z] = q' . kv : (L x MPAD) . (MPAD x DHEAD)  -> reuse kbuf
    wmma_gemm<false, true><<<GRID(SEQ, DHEAD, BH), blk, 0, stream>>>(
        qp, MPAD, (long)SEQ * MPAD, 0, 1,
        kvb, DHEAD, (long)MPAD * DHEAD, 0, 1,
        kbuf, DHEAD, (long)SEQ * DHEAD, 0, 1,
        nullptr, 1.0f, SEQ, DHEAD, MPAD);

    // 11) denominator + normalize + merge heads -> reuse qbuf
    norm_merge_kernel<<<(BH * SEQ) / 8, 256, 0, stream>>>(qp, ksumb, kbuf, qbuf);

    // 12) output projection: (BL x D) . (D x D)^T + out_b
    wmma_gemm<false, false><<<GRID(BL, DIM, 1), blk, 0, stream>>>(
        qbuf, DIM, 0, 0, 1,  out_w, DIM, 0, 0, 1,
        out, DIM, 0, 0, 1,  out_b, 1.0f, BL, DIM, DIM);

    #undef GRID
}
